// MultiHeadAttentionWithFinancialBias_50233937494146
// MI455X (gfx1250) — compile-verified, matrix operand order checked by reading, not run on hardware
//
#include <hip/hip_runtime.h>

// MI455X / gfx1250, wave32, WMMA f16->f32.
// B=8, S=1024, D=1024, H=16, DK=64
#define B_  8
#define S_  1024
#define D_  1024
#define H_  16
#define DK_ 64

typedef _Float16 v16h __attribute__((ext_vector_type(16)));
typedef _Float16 v8h  __attribute__((ext_vector_type(8)));
typedef float    v8f  __attribute__((ext_vector_type(8)));
typedef float    v4f  __attribute__((ext_vector_type(4)));

static __device__ __forceinline__ v8f wmma_f16(v16h a, v16h b, v8f c) {
    return __builtin_amdgcn_wmma_f32_16x16x32_f16(
        /*neg_a=*/false, a, /*neg_b=*/false, b,
        /*c_mod=*/(short)0, c, /*reuse_a=*/false, /*reuse_b=*/false);
}

// ds_swizzle xor-reduction step (group-of-32 mode: offset = xor<<10 | and=0x1f)
template <int IMM>
static __device__ __forceinline__ float dswz(float x) {
    return __int_as_float(__builtin_amdgcn_ds_swizzle(__float_as_int(x), IMM));
}
#define SWZ_X1 0x041F
#define SWZ_X2 0x081F
#define SWZ_X4 0x101F
#define SWZ_X8 0x201F

// ---------------------------------------------------------------------------
// fp32 -> f16 conversion, 8 elements/thread (vectorized b128 in, b128 out)
// ---------------------------------------------------------------------------
__global__ __launch_bounds__(256)
void cvt_f32_to_f16(const float* __restrict__ in, _Float16* __restrict__ out) {
    const size_t i = ((size_t)blockIdx.x * blockDim.x + threadIdx.x) * 8;
    const v4f a = *(const v4f*)(in + i);
    const v4f b = *(const v4f*)(in + i + 4);
    v8h o;
#pragma unroll
    for (int j = 0; j < 4; ++j) { o[j] = (_Float16)a[j]; o[4 + j] = (_Float16)b[j]; }
    *(v8h*)(out + i) = o;
}

// ---------------------------------------------------------------------------
// C[M,N] = A[M,K] @ W[N,K]^T + bias[N]   (A, W are f16; fp32 accumulate)
// OUT_MODE: 0 = f16 row-major [M,N], 1 = fp32 row-major [M,N],
//           2 = f16 transposed per batch: Out[b][n][s], m = b*S_+s  (for V)
// Block: 128 threads = 4 waves; wave tile 32x64 (2 A-frags x 4 B-frags,
// each B-frag feeds 2 WMMAs); block tile 128x64.
// ---------------------------------------------------------------------------
template <int OUT_MODE>
__global__ __launch_bounds__(128)
void gemm_xWT_bias(const _Float16* __restrict__ A,
                   const _Float16* __restrict__ W,
                   const float* __restrict__ bias,
                   void* __restrict__ Out,
                   int N, int K) {
    const int lane  = threadIdx.x & 31;
    const int wave  = threadIdx.x >> 5;
    const int half_ = lane >> 4;
    const int l16   = lane & 15;
    const int rbase = blockIdx.y * 128 + wave * 32;
    const int nbase = blockIdx.x * 64;

    v8f acc[2][4] = {};

    const _Float16* Arow0 = A + (size_t)(rbase + l16) * K;
    const _Float16* Arow1 = Arow0 + (size_t)16 * K;

    for (int kb = 0; kb < K; kb += 32) {
        const int k0 = kb + 8 * half_;
        // two A fragments (16x32 each), b128 loads
        v16h a[2];
        {
            const v8h lo0 = *(const v8h*)(Arow0 + k0);
            const v8h hi0 = *(const v8h*)(Arow0 + k0 + 16);
            const v8h lo1 = *(const v8h*)(Arow1 + k0);
            const v8h hi1 = *(const v8h*)(Arow1 + k0 + 16);
#pragma unroll
            for (int j = 0; j < 8; ++j) {
                a[0][j] = lo0[j]; a[0][8 + j] = hi0[j];
                a[1][j] = lo1[j]; a[1][8 + j] = hi1[j];
            }
        }
#pragma unroll
        for (int nt = 0; nt < 4; ++nt) {
            const _Float16* Wrow = W + (size_t)(nbase + nt * 16 + l16) * K;
            const v8h lo = *(const v8h*)(Wrow + k0);
            const v8h hi = *(const v8h*)(Wrow + k0 + 16);
            v16h bf;
#pragma unroll
            for (int j = 0; j < 8; ++j) { bf[j] = lo[j]; bf[8 + j] = hi[j]; }
            acc[0][nt] = wmma_f16(a[0], bf, acc[0][nt]);
            acc[1][nt] = wmma_f16(a[1], bf, acc[1][nt]);
        }
    }

#pragma unroll
    for (int nt = 0; nt < 4; ++nt) {
        const int n  = nbase + nt * 16 + l16;
        const float bn = bias[n];
#pragma unroll
        for (int mt = 0; mt < 2; ++mt) {
#pragma unroll
            for (int r = 0; r < 8; ++r) {
                const int m = rbase + mt * 16 + 8 * half_ + r;
                const float v = acc[mt][nt][r] + bn;
                if constexpr (OUT_MODE == 0) {
                    ((_Float16*)Out)[(size_t)m * N + n] = (_Float16)v;
                } else if constexpr (OUT_MODE == 1) {
                    ((float*)Out)[(size_t)m * N + n] = v;
                } else {
                    const int b = m >> 10, s = m & (S_ - 1);
                    ((_Float16*)Out)[((size_t)b * D_ + n) * S_ + s] = (_Float16)v;
                }
            }
        }
    }
}

// ---------------------------------------------------------------------------
// Attention core: one wave per (b, h, 16-query tile).
//   scores[16,1024] -> LDS fp32 (64 KB), fused scale/regime-bias/mask
//   softmax via ds_swizzle xor reductions (wave32 C/D layout)
//   attn (fp32) -> global;  probs (f16) -> LDS (32 KB)
//   ctx[16,64] = P @ V via WMMA; A from LDS b128, B from transposed V b128
// ---------------------------------------------------------------------------
__global__ __launch_bounds__(32)
void attention_core(const _Float16* __restrict__ Qh,
                    const _Float16* __restrict__ Kh,
                    const _Float16* __restrict__ Vt,   // [B, D, S] f16
                    const int*      __restrict__ mask,
                    const int*      __restrict__ regime,
                    const float*    __restrict__ regime_emb,
                    float*          __restrict__ attn_out,
                    _Float16*       __restrict__ ctx_out) {
    __shared__ float     sc[16 * S_];   // 64 KB raw scores / exp
    __shared__ _Float16  ph[16 * S_];   // 32 KB normalized probs (f16)

    const int lane  = threadIdx.x & 31;
    const int half_ = lane >> 4;
    const int l16   = lane & 15;
    const int qt = blockIdx.x;
    const int h  = blockIdx.y;
    const int b  = blockIdx.z;
    const int qbase = qt * 16;

    const float scale = 0.125f;                       // 1/sqrt(64)
    const float hbias = regime_emb[regime[b] * H_ + h];

    const size_t bh_off = (size_t)b * S_ * D_ + (size_t)h * DK_;
    const _Float16* Qb = Qh + bh_off;
    const _Float16* Kb = Kh + bh_off;

    // Q fragments: two 16x32 A-frags covering DK=64
    v16h qa[2];
    {
        const _Float16* qrow = Qb + (size_t)(qbase + l16) * D_;
#pragma unroll
        for (int t = 0; t < 2; ++t) {
            const v8h lo = *(const v8h*)(qrow + t * 32 + 8 * half_);
            const v8h hi = *(const v8h*)(qrow + t * 32 + 8 * half_ + 16);
#pragma unroll
            for (int j = 0; j < 8; ++j) { qa[t][j] = lo[j]; qa[t][8 + j] = hi[j]; }
        }
    }

    float rmax[8];
#pragma unroll
    for (int r = 0; r < 8; ++r) rmax[r] = -3.0e38f;

    // ---- pass 1: scores = Q K^T, fused scale/bias/mask, to LDS ----
    for (int kt = 0; kt < S_ / 16; ++kt) {
        v8f accS = {};
        const _Float16* krow = Kb + (size_t)(kt * 16 + l16) * D_;
#pragma unroll
        for (int t = 0; t < 2; ++t) {
            const v8h lo = *(const v8h*)(krow + t * 32 + 8 * half_);
            const v8h hi = *(const v8h*)(krow + t * 32 + 8 * half_ + 16);
            v16h kf;
#pragma unroll
            for (int j = 0; j < 8; ++j) { kf[j] = lo[j]; kf[8 + j] = hi[j]; }
            accS = wmma_f16(qa[t], kf, accS);
        }
        const int col  = kt * 16 + l16;
        const bool keep = mask[(size_t)b * S_ + col] != 0;
#pragma unroll
        for (int r = 0; r < 8; ++r) {
            float v = accS[r] * scale + hbias;
            if (!keep) v = -1.0e9f;
            sc[(8 * half_ + r) * S_ + col] = v;
            rmax[r] = fmaxf(rmax[r], v);
        }
    }
    // row max across the 16 lanes of each half (single ds_swizzle per step)
#pragma unroll
    for (int r = 0; r < 8; ++r) {
        rmax[r] = fmaxf(rmax[r], dswz<SWZ_X1>(rmax[r]));
        rmax[r] = fmaxf(rmax[r], dswz<SWZ_X2>(rmax[r]));
        rmax[r] = fmaxf(rmax[r], dswz<SWZ_X4>(rmax[r]));
        rmax[r] = fmaxf(rmax[r], dswz<SWZ_X8>(rmax[r]));
    }

    // ---- pass 2: exp(s - max) in place, row sums ----
    float rsum[8];
#pragma unroll
    for (int r = 0; r < 8; ++r) rsum[r] = 0.0f;
    for (int c = l16; c < S_; c += 16) {
#pragma unroll
        for (int r = 0; r < 8; ++r) {
            const int row = 8 * half_ + r;
            const float e = __expf(sc[row * S_ + c] - rmax[r]);
            sc[row * S_ + c] = e;
            rsum[r] += e;
        }
    }
#pragma unroll
    for (int r = 0; r < 8; ++r) {
        rsum[r] += dswz<SWZ_X1>(rsum[r]);
        rsum[r] += dswz<SWZ_X2>(rsum[r]);
        rsum[r] += dswz<SWZ_X4>(rsum[r]);
        rsum[r] += dswz<SWZ_X8>(rsum[r]);
    }
    float rinv[8];
#pragma unroll
    for (int r = 0; r < 8; ++r) rinv[r] = 1.0f / rsum[r];

    // ---- pass 3: normalize -> attn (fp32, global) + probs (f16, LDS) ----
    float* attnBase = attn_out + (((size_t)b * H_ + h) * S_ + qbase) * S_;
    for (int c = l16; c < S_; c += 16) {
#pragma unroll
        for (int r = 0; r < 8; ++r) {
            const int row = 8 * half_ + r;
            const float p = sc[row * S_ + c] * rinv[r];
            attnBase[(size_t)row * S_ + c] = p;
            ph[row * S_ + c] = (_Float16)p;
        }
    }

    // ---- pass 4: ctx[16,64] = P[16,1024] @ V[1024,64] ----
    const _Float16* VtB = Vt + ((size_t)b * D_ + h * DK_) * S_;  // rows = dk, stride S_
    v8f accC[4] = {};
    for (int kb2 = 0; kb2 < S_; kb2 += 32) {
        const int k0 = kb2 + 8 * half_;
        // A frag from LDS probs (row = l16), two b128 DS loads
        v16h pa;
        {
            const v8h lo = *(const v8h*)(ph + l16 * S_ + k0);
            const v8h hi = *(const v8h*)(ph + l16 * S_ + k0 + 16);
#pragma unroll
            for (int j = 0; j < 8; ++j) { pa[j] = lo[j]; pa[8 + j] = hi[j]; }
        }
#pragma unroll
        for (int nt = 0; nt < 4; ++nt) {
            // B[k][n] = V[k][dk] = Vt[dk][k] : K-contiguous -> b128 loads
            const _Float16* vrow = VtB + (size_t)(nt * 16 + l16) * S_;
            const v8h lo = *(const v8h*)(vrow + k0);
            const v8h hi = *(const v8h*)(vrow + k0 + 16);
            v16h vf;
#pragma unroll
            for (int j = 0; j < 8; ++j) { vf[j] = lo[j]; vf[8 + j] = hi[j]; }
            accC[nt] = wmma_f16(pa, vf, accC[nt]);
        }
    }
    // store ctx (f16) at [b, q, h*64 + col]  (== bqhd reshape)
    _Float16* ctxBase = ctx_out + bh_off;
#pragma unroll
    for (int nt = 0; nt < 4; ++nt) {
        const int col = nt * 16 + l16;
#pragma unroll
        for (int r = 0; r < 8; ++r) {
            const int row = qbase + 8 * half_ + r;
            ctxBase[(size_t)row * D_ + col] = (_Float16)accC[nt][r];
        }
    }
}

// ---------------------------------------------------------------------------
extern "C" void kernel_launch(void* const* d_in, const int* in_sizes, int n_in,
                              void* d_out, int out_size, void* d_ws, size_t ws_size,
                              hipStream_t stream) {
    const float* query = (const float*)d_in[0];
    const float* key_  = (const float*)d_in[1];
    const float* value = (const float*)d_in[2];
    const int*   mask  = (const int*)d_in[3];
    const int*   regime = (const int*)d_in[4];
    const float* Wq = (const float*)d_in[5];
    const float* bq = (const float*)d_in[6];
    const float* Wk = (const float*)d_in[7];
    const float* bk = (const float*)d_in[8];
    const float* Wv = (const float*)d_in[9];
    const float* bv = (const float*)d_in[10];
    const float* Wo = (const float*)d_in[11];
    const float* bo = (const float*)d_in[12];
    const float* regime_emb = (const float*)d_in[13];

    float* out  = (float*)d_out;                       // [B,S,D] fp32
    float* attn = out + (size_t)B_ * S_ * D_;          // [B,H,S,S] fp32

    // workspace layout (f16): Xh, Qh, Kh, Vt, Ctx (16 MB each) + 4 weights (2 MB each)
    const size_t BSD = (size_t)B_ * S_ * D_;
    const size_t DD  = (size_t)D_ * D_;
    _Float16* Xh  = (_Float16*)d_ws;
    _Float16* Qh  = Xh + BSD;
    _Float16* Kh  = Qh + BSD;
    _Float16* Vt  = Kh + BSD;      // transposed: [B, D, S]
    _Float16* Ctx = Vt + BSD;
    _Float16* WqH = Ctx + BSD;
    _Float16* WkH = WqH + DD;
    _Float16* WvH = WkH + DD;
    _Float16* WoH = WvH + DD;

    const dim3 cblk(256, 1, 1);
    const dim3 cgridW((unsigned)(DD / 8 / 256), 1, 1);
    const dim3 cgridX((unsigned)(BSD / 8 / 256), 1, 1);

    cvt_f32_to_f16<<<cgridW, cblk, 0, stream>>>(Wq, WqH);
    cvt_f32_to_f16<<<cgridW, cblk, 0, stream>>>(Wk, WkH);
    cvt_f32_to_f16<<<cgridW, cblk, 0, stream>>>(Wv, WvH);
    cvt_f32_to_f16<<<cgridW, cblk, 0, stream>>>(Wo, WoH);

    const dim3 gblk(128, 1, 1);
    const dim3 ggrid(D_ / 64, (B_ * S_) / 128, 1);     // (16, 64)

    // Q projection
    cvt_f32_to_f16<<<cgridX, cblk, 0, stream>>>(query, Xh);
    gemm_xWT_bias<0><<<ggrid, gblk, 0, stream>>>(Xh, WqH, bq, Qh, D_, D_);
    // K projection
    cvt_f32_to_f16<<<cgridX, cblk, 0, stream>>>(key_, Xh);
    gemm_xWT_bias<0><<<ggrid, gblk, 0, stream>>>(Xh, WkH, bk, Kh, D_, D_);
    // V projection, output transposed [B, D, S]
    cvt_f32_to_f16<<<cgridX, cblk, 0, stream>>>(value, Xh);
    gemm_xWT_bias<2><<<ggrid, gblk, 0, stream>>>(Xh, WvH, bv, Vt, D_, D_);

    // attention: one wave per (b, h, 16-query tile)
    attention_core<<<dim3(S_ / 16, H_, B_), dim3(32, 1, 1), 0, stream>>>(
        Qh, Kh, Vt, mask, regime, regime_emb, attn, Ctx);

    // output projection: f16 in, fp32 out
    gemm_xWT_bias<1><<<ggrid, gblk, 0, stream>>>(Ctx, WoH, bo, out, D_, D_);
}
